// CompressedBlock_57561151701656
// MI455X (gfx1250) — matrix operand order repaired and checked by reading, not compiled, not run on hardware
//
#include <hip/hip_runtime.h>
#include <hip/hip_bf16.h>

typedef _Float16 v16h __attribute__((ext_vector_type(16)));
typedef _Float16 v8h  __attribute__((ext_vector_type(8)));
typedef float    v8f  __attribute__((ext_vector_type(8)));

#define BB   64
#define TT   1024
#define CC   768
#define TH   512          // tokens per half (even/odd)
#define KK   51           // merged tokens
#define TU   461          // unmerged tokens (TH - KK)
#define TOUT 973          // TU + TH
#define APAD 776          // LDS row stride in halfs (768 + 8) -> conflict-free
#define NKT  (CC / 32)    // 24 K-steps

// ---------------------------------------------------------------------------
// Kernel 1: row-normalize metric, convert to f16, split even/odd tokens.
// ---------------------------------------------------------------------------
__global__ __launch_bounds__(256)
void normalize_kernel(const float* __restrict__ metric,
                      _Float16* __restrict__ mA, _Float16* __restrict__ mB) {
  const int row = blockIdx.x;          // 0 .. B*T-1
  const int b   = row >> 10;
  const int t   = row & 1023;
  const int tid = threadIdx.x;
  const float* p = metric + (size_t)row * CC;

  float v[3];
  float ss = 0.f;
#pragma unroll
  for (int j = 0; j < 3; ++j) { v[j] = p[tid + 256 * j]; ss += v[j] * v[j]; }

  __shared__ float red[8];
#pragma unroll
  for (int off = 16; off >= 1; off >>= 1) ss += __shfl_xor(ss, off, 32);
  if ((tid & 31) == 0) red[tid >> 5] = ss;
  __syncthreads();
  if (tid < 8) {
    float s = red[tid];
#pragma unroll
    for (int off = 4; off >= 1; off >>= 1) s += __shfl_xor(s, off, 32);
    if (tid == 0) red[0] = s;
  }
  __syncthreads();
  const float scale = rsqrtf(red[0]);

  _Float16* dst = ((t & 1) ? mB : mA) + ((size_t)b * TH + (t >> 1)) * CC;
#pragma unroll
  for (int j = 0; j < 3; ++j) dst[tid + 256 * j] = (_Float16)(v[j] * scale);
}

// ---------------------------------------------------------------------------
// Kernel 2: fused WMMA similarity GEMM + row max/argmax.
// Block = (16 'a' rows, batch). 8 waves x 4 N-tiles. A tile in LDS.
// Fully-unrolled K loop with ping-pong prefetch: SSA renaming removes all
// cross-buffer copies (no WAR-hazard nops), first WMMA folds C=0 inline,
// loads of step kt+1 issue before WMMAs of step kt (partial s_wait_loadcnt).
// ---------------------------------------------------------------------------
__global__ __launch_bounds__(256)
void score_argmax_kernel(const _Float16* __restrict__ mA,
                         const _Float16* __restrict__ mB,
                         unsigned long long* __restrict__ nodePacked) {
  const int b       = blockIdx.y;
  const int rowBase = blockIdx.x * 16;
  const int tid     = threadIdx.x;
  const int wave    = tid >> 5;
  const int lane    = tid & 31;
  const int l15     = lane & 15;
  const int hi      = lane >> 4;

  __shared__ _Float16 aTile[16 * APAD];          // ~24.3 KB
  __shared__ unsigned long long smax[16];
  if (tid < 16) smax[tid] = 0ull;

  // Cooperatively stage the 16x768 A tile (1536 x 16B chunks).
  for (int c = tid; c < 16 * (CC / 8); c += 256) {
    const int r   = c / (CC / 8);
    const int col = (c % (CC / 8)) * 8;
    *(v8h*)(aTile + r * APAD + col) =
        *(const v8h*)(mA + ((size_t)b * TH + rowBase + r) * CC + col);
  }
  __syncthreads();

  int ncol[4];
  const _Float16* bRow[4];
#pragma unroll
  for (int t = 0; t < 4; ++t) {
    const int nt = wave + t * 8;                 // this wave's N tiles
    ncol[t] = nt * 16;
    // B fragment: column N = l15; lanes 0-15 hold K=0..15, lanes 16-31 K=16..31
    bRow[t] = mB + ((size_t)b * TH + ncol[t] + l15) * CC + hi * 16;
  }
  const _Float16* aLds = aTile + l15 * APAD;     // this lane's A row in LDS

  union AF { v16h v; v8h h[2]; };
  v8f  acc[4] = {};
  v16h b0[4], b1[4];
  AF   a0, a1;

  // Preload K-step 0 into the even buffers.
#pragma unroll
  for (int t = 0; t < 4; ++t) b0[t] = *(const v16h*)(bRow[t]);
  a0.h[0] = *(const v8h*)(aLds + hi * 16);
  a0.h[1] = *(const v8h*)(aLds + hi * 16 + 8);

#pragma unroll
  for (int kt = 0; kt < NKT; kt += 2) {
    const int kb1 = (kt + 1) * 32;
    const int kb2 = (kt + 2 < NKT) ? (kt + 2) * 32 : 0;   // compile-time tail

    // Prefetch odd step into b1/a1.
#pragma unroll
    for (int t = 0; t < 4; ++t) b1[t] = *(const v16h*)(bRow[t] + kb1);
    a1.h[0] = *(const v8h*)(aLds + kb1 + hi * 16);
    a1.h[1] = *(const v8h*)(aLds + kb1 + hi * 16 + 8);

#pragma unroll
    for (int t = 0; t < 4; ++t)
      acc[t] = __builtin_amdgcn_wmma_f32_16x16x32_f16(
          false, a0.v, false, b0[t], (short)0, acc[t], false, false);

    // Prefetch next even step into b0/a0.
#pragma unroll
    for (int t = 0; t < 4; ++t) b0[t] = *(const v16h*)(bRow[t] + kb2);
    a0.h[0] = *(const v8h*)(aLds + kb2 + hi * 16);
    a0.h[1] = *(const v8h*)(aLds + kb2 + hi * 16 + 8);

#pragma unroll
    for (int t = 0; t < 4; ++t)
      acc[t] = __builtin_amdgcn_wmma_f32_16x16x32_f16(
          false, a1.v, false, b1[t], (short)0, acc[t], false, false);
  }

  // In-register max/argmax. acc[i] at this lane is element (M=i+hi*8, N=l15+ncol).
  float runMax[8];
  int   runIdx[8];
#pragma unroll
  for (int i = 0; i < 8; ++i) { runMax[i] = -3.4e38f; runIdx[i] = 0; }
#pragma unroll
  for (int t = 0; t < 4; ++t) {
#pragma unroll
    for (int i = 0; i < 8; ++i) {
      const float v = acc[t][i];
      if (v > runMax[i]) { runMax[i] = v; runIdx[i] = ncol[t] + l15; }
    }
  }
  // reduce across the 16 lanes that share a row (xor masks stay inside halves)
#pragma unroll
  for (int off = 8; off >= 1; off >>= 1) {
#pragma unroll
    for (int i = 0; i < 8; ++i) {
      const float om = __shfl_xor(runMax[i], off, 32);
      const int   oi = __shfl_xor(runIdx[i], off, 32);
      if (om > runMax[i] || (om == runMax[i] && oi < runIdx[i])) {
        runMax[i] = om; runIdx[i] = oi;
      }
    }
  }
  if (l15 == 0) {
#pragma unroll
    for (int i = 0; i < 8; ++i) {
      const int row = i + hi * 8;
      unsigned u = __float_as_uint(runMax[i]);
      u = (u & 0x80000000u) ? ~u : (u | 0x80000000u);   // order-preserving
      const unsigned long long key =
          ((unsigned long long)u << 32) | (unsigned)runIdx[i];
      atomicMax(&smax[row], key);                        // ds_max_u64
    }
  }
  __syncthreads();
  if (tid < 16) nodePacked[(size_t)b * TH + rowBase + tid] = smax[tid];
}

// ---------------------------------------------------------------------------
// Kernel 3: per-batch bitonic argsort (descending by max, stable by index).
// ---------------------------------------------------------------------------
__global__ __launch_bounds__(512)
void sort_kernel(const unsigned long long* __restrict__ nodePacked,
                 int* __restrict__ srcIdx, int* __restrict__ dstIdx,
                 int* __restrict__ unmIdx) {
  const int b   = blockIdx.x;
  const int tid = threadIdx.x;
  __shared__ unsigned long long key[TH];

  const unsigned long long np = nodePacked[(size_t)b * TH + tid];
  const unsigned ord = (unsigned)(np >> 32);
  // ascending sort of (~ord, tid)  ==  descending by score, ties: low index
  key[tid] = ((unsigned long long)(~ord) << 32) | (unsigned)tid;
  __syncthreads();

  for (int k2 = 2; k2 <= TH; k2 <<= 1) {
    for (int j = k2 >> 1; j >= 1; j >>= 1) {
      const int ixj = tid ^ j;
      if (ixj > tid) {
        const bool up = ((tid & k2) == 0);
        const unsigned long long a = key[tid], c = key[ixj];
        if ((a > c) == up) { key[tid] = c; key[ixj] = a; }
      }
      __syncthreads();
    }
  }

  const int row = (int)(key[tid] & 0xFFFFFFFFull);
  if (tid < KK) {
    srcIdx[b * KK + tid] = row;
    dstIdx[b * KK + tid] = (int)(nodePacked[(size_t)b * TH + row] & 0xFFFFFFFFull);
  } else {
    unmIdx[b * TU + (tid - KK)] = row;
  }
}

// ---------------------------------------------------------------------------
// Kernel 4: merge. One block per output row (973 per batch), 768 floats.
// ---------------------------------------------------------------------------
__global__ __launch_bounds__(256)
void merge_kernel(const float* __restrict__ x,
                  const int* __restrict__ srcIdx, const int* __restrict__ dstIdx,
                  const int* __restrict__ unmIdx, float* __restrict__ out) {
  const int r   = blockIdx.x;    // 0 .. 972
  const int b   = blockIdx.y;
  const int tid = threadIdx.x;
  float* op = out + ((size_t)b * TOUT + r) * CC;

  if (r < TU) {                                        // unmerged: pure gather
    const int s = unmIdx[b * TU + r];
    const float* ip = x + ((size_t)b * TT + 2 * s) * CC;
#pragma unroll
    for (int j = 0; j < 3; ++j) op[tid + 256 * j] = ip[tid + 256 * j];
  } else {                                             // dst token + merged srcs
    const int n = r - TU;
    const float* ip = x + ((size_t)b * TT + 2 * n + 1) * CC;
    float acc[3];
#pragma unroll
    for (int j = 0; j < 3; ++j) acc[j] = ip[tid + 256 * j];
    int count = 1;
    for (int jj = 0; jj < KK; ++jj) {                  // uniform scalar loads
      if (dstIdx[b * KK + jj] == n) {
        const float* sp = x + ((size_t)b * TT + 2 * srcIdx[b * KK + jj]) * CC;
#pragma unroll
        for (int j = 0; j < 3; ++j) acc[j] += sp[tid + 256 * j];
        ++count;
      }
    }
    const float inv = 1.0f / (float)count;
#pragma unroll
    for (int j = 0; j < 3; ++j) op[tid + 256 * j] = acc[j] * inv;
  }
}

// ---------------------------------------------------------------------------
extern "C" void kernel_launch(void* const* d_in, const int* in_sizes, int n_in,
                              void* d_out, int out_size, void* d_ws, size_t ws_size,
                              hipStream_t stream) {
  (void)in_sizes; (void)n_in; (void)out_size; (void)ws_size;
  const float* metric = (const float*)d_in[0];
  const float* x      = (const float*)d_in[1];
  float* out          = (float*)d_out;

  char* ws = (char*)d_ws;
  const size_t SA = (size_t)BB * TH * CC * sizeof(_Float16);   // 50.3 MB each
  _Float16* mA = (_Float16*)ws;
  _Float16* mB = (_Float16*)(ws + SA);
  unsigned long long* nodePacked = (unsigned long long*)(ws + 2 * SA);
  int* srcIdx = (int*)(ws + 2 * SA + (size_t)BB * TH * 8);
  int* dstIdx = srcIdx + BB * KK;
  int* unmIdx = dstIdx + BB * KK;

  normalize_kernel<<<dim3(BB * TT), 256, 0, stream>>>(metric, mA, mB);
  score_argmax_kernel<<<dim3(TH / 16, BB), 256, 0, stream>>>(mA, mB, nodePacked);
  sort_kernel<<<dim3(BB), TH, 0, stream>>>(nodePacked, srcIdx, dstIdx, unmIdx);
  merge_kernel<<<dim3(TOUT, BB), 256, 0, stream>>>(x, srcIdx, dstIdx, unmIdx, out);
}
